// MultiHeadAttention_87247965651527
// MI455X (gfx1250) — matrix operand order repaired and checked by reading, not compiled, not run on hardware
//
#include <hip/hip_runtime.h>
#include <hip/hip_bf16.h>
#include <stdint.h>

typedef __attribute__((ext_vector_type(16))) __bf16 v16bf;
typedef __attribute__((ext_vector_type(8)))  float  v8f;

// Problem constants (match reference)
constexpr int Bc = 2;
constexpr int Sc = 2048;
constexpr int Ec = 1024;
constexpr int Hc = 16;
constexpr int Dc = 64;

// ---------------------------------------------------------------------------
// Fragment loaders. 16-bit WMMA operand k order used here (contiguous, same for A and B):
//   A (16x32): lane m holds row m; elems e=0..15 -> k = 16*half + e
//   B (32x16): lane n holds col n; elems e=0..15 -> k = 16*half + e
// Both are k-contiguous runs => each fragment is exactly two 16B loads.
// ---------------------------------------------------------------------------
__device__ __forceinline__ v16bf load_afrag(const __bf16* rowk, int half) {
    union { v16bf v; float4 q[2]; } u;
    u.q[0] = *reinterpret_cast<const float4*>(rowk + 16 * half);
    u.q[1] = *reinterpret_cast<const float4*>(rowk + 16 * half + 8);
    return u.v;
}
__device__ __forceinline__ v16bf load_bfrag(const __bf16* colk, int half) {
    union { v16bf v; float4 q[2]; } u;
    u.q[0] = *reinterpret_cast<const float4*>(colk + 16 * half);
    u.q[1] = *reinterpret_cast<const float4*>(colk + 16 * half + 8);
    return u.v;
}

// ---------------------------------------------------------------------------
// Async global -> LDS 16B copy (CDNA5, ASYNCcnt). Per-lane LDS scatter:
// each lane supplies its own LDS byte offset and global address, so the
// padded-row LDS layout is preserved with no VGPR round trip.
// ---------------------------------------------------------------------------
__device__ __forceinline__ void async_b128(void* lds, const void* g) {
    const uint32_t l = (uint32_t)(uintptr_t)lds;   // low 32 bits = LDS offset
    asm volatile("global_load_async_to_lds_b128 %0, %1, off"
                 :: "v"(l), "v"(g) : "memory");
}
__device__ __forceinline__ void wait_async() {
    asm volatile("s_wait_asynccnt 0x0" ::: "memory");
}

// ---------------------------------------------------------------------------
// Split fp32 -> (hi, lo) bf16 pair, row-major (A-side operands).
// ---------------------------------------------------------------------------
__global__ void split_bf16_kernel(const float* __restrict__ in,
                                  __bf16* __restrict__ hi,
                                  __bf16* __restrict__ lo, int n)
{
    const int i = blockIdx.x * blockDim.x + threadIdx.x;
    if (i >= n) return;
    const float x = in[i];
    const __bf16 h = (__bf16)x;
    hi[i] = h;
    lo[i] = (__bf16)(x - (float)h);
}

// ---------------------------------------------------------------------------
// Split fp32 [K,N] -> transposed (hi, lo) bf16 [N,K]  (B-side operands:
// k-contiguous rows so GEMM can stage them with async b128 copies).
// ---------------------------------------------------------------------------
__global__ void split_bf16_t_kernel(const float* __restrict__ in,
                                    __bf16* __restrict__ hiT,
                                    __bf16* __restrict__ loT,
                                    int Kdim, int Ndim)
{
    const int i = blockIdx.x * blockDim.x + threadIdx.x;
    if (i >= Kdim * Ndim) return;
    const int n = i % Ndim;
    const int k = i / Ndim;
    const float x = in[i];
    const __bf16 h = (__bf16)x;
    hiT[(size_t)n * Kdim + k] = h;
    loT[(size_t)n * Kdim + k] = (__bf16)(x - (float)h);
}

// ---------------------------------------------------------------------------
// Split-precision GEMM: C[M,N] (fp32) = (Ah+Al)[M,K] @ (Bh+Bl)[K,N]
// with B supplied transposed [N,K]. 3 bf16 WMMAs per tile (AhBh+AhBl+AlBh),
// fp32 accumulators. Block = 256 threads (8 waves), 128x64 C tile, K-slab 32,
// double-buffered LDS filled by async global->LDS b128 copies.
// Rows padded to 40 bf16 (80 B: 16B-aligned, conflict-free reads).
// ---------------------------------------------------------------------------
__global__ __launch_bounds__(256) void gemm_bf16x3(
    const __bf16* __restrict__ Ah,  const __bf16* __restrict__ Al,
    const __bf16* __restrict__ Bth, const __bf16* __restrict__ Btl,
    float* __restrict__ C, int M, int N, int K)
{
    __shared__ __bf16 Ahs[2][128][40];
    __shared__ __bf16 Als[2][128][40];
    __shared__ __bf16 Bhs[2][64][40];   // [n][k]
    __shared__ __bf16 Bls[2][64][40];

    const int t    = threadIdx.x;
    const int wave = t >> 5;
    const int lane = t & 31;
    const int half = lane >> 4;
    const int ln   = lane & 15;

    const int row0  = blockIdx.x * 128;
    const int col0  = blockIdx.y * 64;
    const int rows0 = (wave & 3) * 32;   // wave's 32-row strip
    const int cols0 = (wave >> 2) * 32;  // wave's 32-col strip

    v8f acc[2][2] = {};

    // async staging of one K=32 slab into buffer `buf`
    auto stage = [&](int buf, int k0) {
        #pragma unroll
        for (int rep = 0; rep < 2; ++rep) {
            const int c  = t + rep * 256;      // 0..511
            const int r  = c >> 2;             // 0..127
            const int kc = (c & 3) * 8;        // 0,8,16,24
            const size_t go = (size_t)(row0 + r) * K + k0 + kc;
            async_b128(&Ahs[buf][r][kc], Ah + go);
            async_b128(&Als[buf][r][kc], Al + go);
        }
        {
            const int n  = t >> 2;             // 0..63
            const int kc = (t & 3) * 8;        // 0,8,16,24
            const size_t go = (size_t)(col0 + n) * K + k0 + kc;
            async_b128(&Bhs[buf][n][kc], Bth + go);
            async_b128(&Bls[buf][n][kc], Btl + go);
        }
    };

    stage(0, 0);
    wait_async();
    __syncthreads();

    for (int k0 = 0; k0 < K; k0 += 32) {
        const int cur = (k0 >> 5) & 1;
        if (k0 + 32 < K) stage(cur ^ 1, k0 + 32);   // overlap copy with WMMA
        if (k0 + 64 < K)
            __builtin_prefetch(Ah + (size_t)(row0 + (t >> 2)) * K + k0 + 64, 0, 0);

        // ---- fragments: two 16B LDS reads each
        v16bf ah[2], al[2], bh[2], bl[2];
        #pragma unroll
        for (int ri = 0; ri < 2; ++ri) {
            ah[ri] = load_afrag(&Ahs[cur][rows0 + ri * 16 + ln][0], half);
            al[ri] = load_afrag(&Als[cur][rows0 + ri * 16 + ln][0], half);
        }
        #pragma unroll
        for (int ci = 0; ci < 2; ++ci) {
            bh[ci] = load_bfrag(&Bhs[cur][cols0 + ci * 16 + ln][0], half);
            bl[ci] = load_bfrag(&Bls[cur][cols0 + ci * 16 + ln][0], half);
        }
        #pragma unroll
        for (int ri = 0; ri < 2; ++ri)
            #pragma unroll
            for (int ci = 0; ci < 2; ++ci) {
                acc[ri][ci] = __builtin_amdgcn_wmma_f32_16x16x32_bf16(
                    false, ah[ri], false, bh[ci], (short)0, acc[ri][ci], false, false);
                acc[ri][ci] = __builtin_amdgcn_wmma_f32_16x16x32_bf16(
                    false, ah[ri], false, bl[ci], (short)0, acc[ri][ci], false, false);
                acc[ri][ci] = __builtin_amdgcn_wmma_f32_16x16x32_bf16(
                    false, al[ri], false, bh[ci], (short)0, acc[ri][ci], false, false);
            }

        wait_async();        // next slab's async copies complete
        __syncthreads();
    }

    #pragma unroll
    for (int ri = 0; ri < 2; ++ri)
        #pragma unroll
        for (int ci = 0; ci < 2; ++ci)
            #pragma unroll
            for (int r = 0; r < 8; ++r) {
                const int row = row0 + rows0 + ri * 16 + r + 8 * half;
                const int col = col0 + cols0 + ci * 16 + ln;
                C[(size_t)row * N + col] = acc[ri][ci][r];
            }
}

// ---------------------------------------------------------------------------
// RoPE + split qkv[B,S,3E] (fp32) -> qr/kr/vr in [B,H,S,D] as bf16.
// 1/sqrt(D) folded into q before rounding.
// ---------------------------------------------------------------------------
__global__ void rope_split_kernel(const float* __restrict__ qkv,
                                  __bf16* __restrict__ qr,
                                  __bf16* __restrict__ kr,
                                  __bf16* __restrict__ vr)
{
    const int idx = blockIdx.x * blockDim.x + threadIdx.x;   // over B*S*E
    if (idx >= Bc * Sc * Ec) return;
    const int d = idx % Dc;
    const int h = (idx / Dc) % Hc;
    const int s = (idx / Ec) % Sc;
    const int b = idx / (Ec * Sc);

    const size_t rowbase = ((size_t)b * Sc + s) * (3 * Ec);
    const int col = h * Dc + d;

    const float q = qkv[rowbase + col];
    const float k = qkv[rowbase + Ec + col];
    const float v = qkv[rowbase + 2 * Ec + col];

    const int i = d & (Dc / 2 - 1);
    const float inv = __powf(10000.0f, -(float)(2 * i) / (float)Dc);
    const float ang = (float)s * inv;
    const float cs = __cosf(ang);
    const float sn = __sinf(ang);

    float qo, ko;
    if (d < Dc / 2) {
        const float q2 = qkv[rowbase + col + Dc / 2];
        const float k2 = qkv[rowbase + Ec + col + Dc / 2];
        qo = q * cs - q2 * sn;
        ko = k * cs - k2 * sn;
    } else {
        const float q2 = qkv[rowbase + col - Dc / 2];
        const float k2 = qkv[rowbase + Ec + col - Dc / 2];
        qo = q * cs + q2 * sn;
        ko = k * cs + k2 * sn;
    }

    const size_t o = (((size_t)b * Hc + h) * Sc + s) * Dc + d;
    qr[o] = (__bf16)(qo * 0.125f);
    kr[o] = (__bf16)ko;
    vr[o] = (__bf16)v;
}

// ---------------------------------------------------------------------------
// Causal flash attention, bf16 WMMA (fp32 accumulate / fp32 softmax).
// One wave per 16-query tile; 8 waves per block. Keys processed 32 at a time:
//   scores: 2 tiles x 2 WMMA k-steps (D=64); P@V: 4 WMMAs (K=32).
// ---------------------------------------------------------------------------
__global__ __launch_bounds__(256) void flash_attn_kernel(
    const __bf16* __restrict__ qr, const __bf16* __restrict__ kr,
    const __bf16* __restrict__ vr, float* __restrict__ out)
{
    __shared__ __bf16 Plds[8][16][40];   // per-wave 16x32 P tile, padded rows

    const int lane = threadIdx.x & 31;
    const int wave = threadIdx.x >> 5;
    const int half = lane >> 4;
    const int ln   = lane & 15;

    const int qt = blockIdx.x * 8 + wave;    // query tile 0..S/16-1
    const int bh = blockIdx.y;
    const int b  = bh >> 4;
    const int h  = bh & 15;
    const size_t base = (size_t)bh * Sc * Dc;

    const __bf16* qrow = qr + (base + (size_t)(qt * 16 + ln) * Dc);
    const v16bf qf0 = load_afrag(qrow, half);
    const v16bf qf1 = load_afrag(qrow + 32, half);

    float mrow[8], lrow[8];
    #pragma unroll
    for (int r = 0; r < 8; ++r) { mrow[r] = -3.0e38f; lrow[r] = 0.0f; }
    v8f acc[4] = {};

    const int jmax = (qt * 16 + 15) >> 5;    // last 32-key block index
    for (int j = 0; j <= jmax; ++j) {
        const int kb = j * 32;

        // ---- scores: two 16x16 tiles (keys kb..kb+15, kb+16..kb+31)
        v8f sc[2] = {};
        #pragma unroll
        for (int c = 0; c < 2; ++c) {
            const __bf16* krow = kr + (base + (size_t)(kb + c * 16 + ln) * Dc);
            const v16bf kf0 = load_bfrag(krow, half);
            const v16bf kf1 = load_bfrag(krow + 32, half);
            sc[c] = __builtin_amdgcn_wmma_f32_16x16x32_bf16(
                false, qf0, false, kf0, (short)0, sc[c], false, false);
            sc[c] = __builtin_amdgcn_wmma_f32_16x16x32_bf16(
                false, qf1, false, kf1, (short)0, sc[c], false, false);
        }

        // ---- online softmax over the 32 columns
        const bool edge = (j == jmax);
        #pragma unroll
        for (int r = 0; r < 8; ++r) {
            const int qrow_g = qt * 16 + r + 8 * half;
            float s0 = sc[0][r];
            float s1 = sc[1][r];
            if (edge) {
                if (kb + ln > qrow_g)      s0 = -3.0e38f;
                if (kb + 16 + ln > qrow_g) s1 = -3.0e38f;
            }
            float x = fmaxf(s0, s1);
            x = fmaxf(x, __shfl_xor(x, 1, 32));
            x = fmaxf(x, __shfl_xor(x, 2, 32));
            x = fmaxf(x, __shfl_xor(x, 4, 32));
            x = fmaxf(x, __shfl_xor(x, 8, 32));
            const float mnew  = fmaxf(mrow[r], x);
            const float alpha = __expf(mrow[r] - mnew);
            const float p0 = __expf(s0 - mnew);
            const float p1 = __expf(s1 - mnew);
            float psum = p0 + p1;
            psum += __shfl_xor(psum, 1, 32);
            psum += __shfl_xor(psum, 2, 32);
            psum += __shfl_xor(psum, 4, 32);
            psum += __shfl_xor(psum, 8, 32);
            lrow[r] = lrow[r] * alpha + psum;
            mrow[r] = mnew;
            Plds[wave][r + 8 * half][ln]      = (__bf16)p0;
            Plds[wave][r + 8 * half][16 + ln] = (__bf16)p1;
            #pragma unroll
            for (int t2 = 0; t2 < 4; ++t2) acc[t2][r] *= alpha;
        }
        asm volatile("s_wait_dscnt 0x0" ::: "memory");

        // ---- O += P @ V  (K = 32 keys, one WMMA per 16-wide D chunk)
        const v16bf pa = load_afrag(&Plds[wave][ln][0], half);
        const __bf16* vrow = vr + (base + (size_t)kb * Dc);
        #pragma unroll
        for (int t2 = 0; t2 < 4; ++t2) {
            v16bf vf;
            #pragma unroll
            for (int e = 0; e < 16; ++e) {
                const int keyloc = 16 * half + e;
                vf[e] = vrow[(size_t)keyloc * Dc + t2 * 16 + ln];
            }
            acc[t2] = __builtin_amdgcn_wmma_f32_16x16x32_bf16(
                false, pa, false, vf, (short)0, acc[t2], false, false);
        }
    }

    // epilogue: normalize, scatter to [B,S,E] (fp32)
    #pragma unroll
    for (int r = 0; r < 8; ++r) {
        const int srow = qt * 16 + r + 8 * half;
        const float invl = 1.0f / lrow[r];
        #pragma unroll
        for (int t2 = 0; t2 < 4; ++t2) {
            out[((size_t)b * Sc + srow) * Ec + h * Dc + t2 * 16 + ln] =
                acc[t2][r] * invl;
        }
    }
}

// ---------------------------------------------------------------------------
// Launch wrapper
// ---------------------------------------------------------------------------
extern "C" void kernel_launch(void* const* d_in, const int* in_sizes, int n_in,
                              void* d_out, int out_size, void* d_ws, size_t ws_size,
                              hipStream_t stream)
{
    const float* x     = (const float*)d_in[0];   // [B,S,E]
    const float* w_qkv = (const float*)d_in[1];   // [E, 3E]
    const float* w_out = (const float*)d_in[2];   // [E, E]
    float* out = (float*)d_out;                   // [B,S,E]

    const int M = Bc * Sc;                        // 4096
    const size_t n_x    = (size_t)M * Ec;
    const size_t n_wqkv = (size_t)Ec * 3 * Ec;
    const size_t n_wout = (size_t)Ec * Ec;
    const size_t n_qkv  = (size_t)M * 3 * Ec;
    const size_t n_bhsd = (size_t)Bc * Hc * Sc * Dc;

    char* ws = (char*)d_ws;
    float*  qkv  = (float*)ws;            ws += n_qkv * 4;   // also reused as attn
    __bf16* xh   = (__bf16*)ws;           ws += n_x * 2;
    __bf16* xl   = (__bf16*)ws;           ws += n_x * 2;
    __bf16* wqh  = (__bf16*)ws;           ws += n_wqkv * 2;  // transposed [3E, E]
    __bf16* wql  = (__bf16*)ws;           ws += n_wqkv * 2;
    __bf16* woh  = (__bf16*)ws;           ws += n_wout * 2;  // transposed [E, E]
    __bf16* wol  = (__bf16*)ws;           ws += n_wout * 2;
    __bf16* qr   = (__bf16*)ws;           ws += n_bhsd * 2;
    __bf16* kr   = (__bf16*)ws;           ws += n_bhsd * 2;
    __bf16* vr   = (__bf16*)ws;           ws += n_bhsd * 2;
    __bf16* ah   = (__bf16*)ws;           ws += n_x * 2;
    __bf16* al   = (__bf16*)ws;           ws += n_x * 2;
    float* attn = qkv;   // qkv buffer is dead after rope_split

    // 0) split inputs (row-major) and weights (transposed) into bf16 hi/lo
    split_bf16_kernel<<<(int)((n_x + 255) / 256), 256, 0, stream>>>(x, xh, xl, (int)n_x);
    split_bf16_t_kernel<<<(int)((n_wqkv + 255) / 256), 256, 0, stream>>>(
        w_qkv, wqh, wql, Ec, 3 * Ec);
    split_bf16_t_kernel<<<(int)((n_wout + 255) / 256), 256, 0, stream>>>(
        w_out, woh, wol, Ec, Ec);

    // 1) QKV projection: [4096,1024] @ [1024,3072] -> fp32
    {
        dim3 grid(M / 128, (3 * Ec) / 64);
        gemm_bf16x3<<<grid, 256, 0, stream>>>(xh, xl, wqh, wql, qkv, M, 3 * Ec, Ec);
    }
    // 2) RoPE + split/transpose to [B,H,S,D] bf16
    {
        const int n = Bc * Sc * Ec;
        rope_split_kernel<<<(n + 255) / 256, 256, 0, stream>>>(qkv, qr, kr, vr);
    }
    // 3) causal flash attention -> attn fp32 [B,S,E]
    {
        dim3 grid((Sc / 16) / 8, Bc * Hc);
        flash_attn_kernel<<<grid, 256, 0, stream>>>(qr, kr, vr, attn);
    }
    // 4) split attention output, then out projection -> out
    split_bf16_kernel<<<(int)((n_x + 255) / 256), 256, 0, stream>>>(attn, ah, al, (int)n_x);
    {
        dim3 grid(M / 128, Ec / 64);
        gemm_bf16x3<<<grid, 256, 0, stream>>>(ah, al, woh, wol, out, M, Ec, Ec);
    }
}
